// RETAIN_10969346474754
// MI455X (gfx1250) — matrix operand order repaired and checked by reading, not compiled, not run on hardware
//
#include <hip/hip_runtime.h>
#include <math.h>

// ---------------------------------------------------------------------------
// Types for CDNA5 WMMA (gfx1250, wave32)
// ---------------------------------------------------------------------------
typedef __bf16 bf16;
typedef __bf16 v16bf __attribute__((ext_vector_type(16)));
typedef __bf16 v8bf  __attribute__((ext_vector_type(8)));
typedef float  v8f   __attribute__((ext_vector_type(8)));

#define WMMA_BF16(a, b, c) \
    __builtin_amdgcn_wmma_f32_16x16x32_bf16(false, (a), false, (b), (short)0, (c), false, false)

// ---------------------------------------------------------------------------
// Fragment loaders (16-bit A 16x32 and B 32x16 VGPR layouts per CDNA5 ISA 7.12.2)
// ---------------------------------------------------------------------------
__device__ inline v16bf load_fragA(const bf16* __restrict__ tile, int ld) {
    int lane = threadIdx.x & 31;
    int row  = lane & 15;
    int koff = (lane & 16) ? 8 : 0;
    const bf16* p = tile + (size_t)row * ld + koff;
    union { v16bf v; v8bf h[2]; } u;
    u.h[0] = *(const v8bf*)(p);
    u.h[1] = *(const v8bf*)(p + 16);
    return u.v;
}

__device__ inline v16bf load_fragB(const bf16* __restrict__ W, int n0, int k0, int ld) {
    int lane = threadIdx.x & 31;
    int col  = lane & 15;
    int kb   = k0 + ((lane & 16) ? 8 : 0);
    const bf16* p = W + (size_t)(n0 + col) * ld + kb;
    union { v16bf v; v8bf h[2]; } u;
    u.h[0] = *(const v8bf*)(p);
    u.h[1] = *(const v8bf*)(p + 16);
    return u.v;
}

// ---------------------------------------------------------------------------
// f32 -> bf16 conversion
// ---------------------------------------------------------------------------
__global__ void cvt_bf16_kernel(const float* __restrict__ in, bf16* __restrict__ out, int n) {
    int i = blockIdx.x * 256 + threadIdx.x;
    if (i < n) out[i] = (bf16)in[i];
}

// ---------------------------------------------------------------------------
// Generic WMMA GEMM:  C[m,n] = sum_k A[m,k] * W[n,k] + bias[n]
// One wave computes a 16x64 strip (4 accumulators): each A fragment feeds 4 WMMAs.
//   MODE 0: outF = C (f32) and outB = C (bf16)          (projection xp)
//   MODE 1: outB = C (bf16)                             (xW precompute)
//   MODE 2: outF = tanh(C) * mulF (fused beta*xp = v)
// ---------------------------------------------------------------------------
template <int MODE>
__global__ __launch_bounds__(32) void wmma_gemm_kernel(
    const bf16* __restrict__ A, const bf16* __restrict__ W,
    const float* __restrict__ bias, float* __restrict__ outF,
    bf16* __restrict__ outB, const float* __restrict__ mulF,
    int N, int K)
{
    int m0 = blockIdx.x * 16;
    int n0 = blockIdx.y * 64;

    const v8f z8 = {0.f, 0.f, 0.f, 0.f, 0.f, 0.f, 0.f, 0.f};
    v8f acc[4] = {z8, z8, z8, z8};
    for (int k0 = 0; k0 < K; k0 += 32) {
        v16bf a = load_fragA(A + (size_t)m0 * K + k0, K);
#pragma unroll
        for (int j = 0; j < 4; j++) {
            v16bf b = load_fragB(W, n0 + j * 16, k0, K);
            acc[j] = WMMA_BF16(a, b, acc[j]);
        }
    }

    int lane = threadIdx.x & 31;
    int cN   = lane & 15;
    int mb   = m0 + ((lane & 16) ? 8 : 0);
#pragma unroll
    for (int j = 0; j < 4; j++) {
        int col  = n0 + j * 16 + cN;
        float bv = bias[col];
#pragma unroll
        for (int i = 0; i < 8; i++) {
            size_t idx = (size_t)(mb + i) * N + col;
            float r = acc[j][i] + bv;
            if (MODE == 0) { outF[idx] = r; outB[idx] = (bf16)r; }
            else if (MODE == 1) { outB[idx] = (bf16)r; }
            else { outF[idx] = tanhf(r) * mulF[idx]; }
        }
    }
}

// ---------------------------------------------------------------------------
// GRU sequential scan. One workgroup (512 threads = 16 waves) owns 16 batch
// rows; h lives in LDS. Wave w owns one 16-column tile of H for all 3 gates,
// so its Whh B-fragments (3 gates x 8 k-tiles = 24 frags = 192 VGPRs) stay
// REGISTER-RESIDENT across all T steps -- no reloads, no spills.
//   xW : [B,T,768] bf16 (input projections, bias already added)
//   Whh: [768,256] bf16, bhh: [768] f32
//   hs : [B,T,256] bf16
// ---------------------------------------------------------------------------
__global__ __launch_bounds__(512) void gru_scan_kernel(
    const bf16* __restrict__ xW, const bf16* __restrict__ Whh,
    const float* __restrict__ bhh, bf16* __restrict__ hs, int T)
{
    __shared__ bf16 hbuf[16][264];   // 16 rows x 256 (+8 pad) bf16
    for (int i = threadIdx.x; i < 16 * 264; i += 512) (&hbuf[0][0])[i] = (bf16)0.f;
    __syncthreads();

    const int wave = threadIdx.x >> 5;          // 0..15 -> column tile
    const int lane = threadIdx.x & 31;
    const int cN   = lane & 15;
    const int mh   = (lane & 16) ? 8 : 0;
    const int colb = wave * 16;
    const int col  = colb + cN;
    const int bbase = blockIdx.x * 16;

    // register-resident Whh fragments: [k-tile][gate]
    v16bf bf[8][3];
#pragma unroll
    for (int kt = 0; kt < 8; kt++)
#pragma unroll
        for (int g = 0; g < 3; g++)
            bf[kt][g] = load_fragB(Whh, g * 256 + colb, kt * 32, 256);

    // hoisted hidden biases
    float bh[3];
#pragma unroll
    for (int g = 0; g < 3; g++) bh[g] = bhh[g * 256 + col];

    const v8f z8 = {0.f, 0.f, 0.f, 0.f, 0.f, 0.f, 0.f, 0.f};

    for (int t = 0; t < T; t++) {
        v8f acc[3] = {z8, z8, z8};
#pragma unroll
        for (int kt = 0; kt < 8; kt++) {
            // A fragment of h from LDS, transient (one live at a time)
            const bf16* p = &hbuf[cN][kt * 32 + ((lane & 16) ? 8 : 0)];
            union { v16bf v; v8bf h[2]; } u;
            u.h[0] = *(const v8bf*)(p);
            u.h[1] = *(const v8bf*)(p + 16);
#pragma unroll
            for (int g = 0; g < 3; g++)
                acc[g] = WMMA_BF16(u.v, bf[kt][g], acc[g]);
        }

        // gate math on accumulator layout: element i -> (M = mh+i, N = col)
        float hnew[8];
#pragma unroll
        for (int i = 0; i < 8; i++) {
            int m = mh + i;
            size_t row = (size_t)(bbase + m) * T + t;
            const bf16* xwp = xW + row * 768;
            float xr = (float)xwp[col];
            float xz = (float)xwp[256 + col];
            float xn = (float)xwp[512 + col];
            if (t + 1 < T) __builtin_prefetch(xwp + 768 + col, 0, 1);  // next step's gate inputs
            float hr = acc[0][i] + bh[0];
            float hz = acc[1][i] + bh[1];
            float hn = acc[2][i] + bh[2];
            float hp = (float)hbuf[m][col];
            float r = 1.f / (1.f + __expf(-(xr + hr)));
            float z = 1.f / (1.f + __expf(-(xz + hz)));
            float n = tanhf(xn + r * hn);
            float hv = (1.f - z) * n + z * hp;
            hnew[i] = hv;
            hs[row * 256 + col] = (bf16)hv;
        }
        __syncthreads();   // everyone done reading h_t
#pragma unroll
        for (int i = 0; i < 8; i++)
            hbuf[mh + i][col] = (bf16)hnew[i];
        __syncthreads();   // h_{t+1} visible
    }
}

// ---------------------------------------------------------------------------
// scores[b,t] = xa[b,t,:] . Wa + ba    (one thread per (b,t))
// ---------------------------------------------------------------------------
__global__ __launch_bounds__(256) void scores_kernel(
    const bf16* __restrict__ xa, const float* __restrict__ Wa,
    const float* __restrict__ ba, float* __restrict__ sc)
{
    int i = blockIdx.x * 256 + threadIdx.x;
    const bf16* p = xa + (size_t)i * 256;
    float s = ba[0];
#pragma unroll 8
    for (int k = 0; k < 256; k++) s += (float)p[k] * Wa[k];
    sc[i] = s;
}

// ---------------------------------------------------------------------------
// Causal attention == online-softmax prefix scan over T (scores depend only
// on source index):  out[b,t,h] = (sum_{s<=t} e^{sc_s} v[b,s,h]) / sum e^{sc_s}
// ---------------------------------------------------------------------------
__global__ __launch_bounds__(256) void attn_scan_kernel(
    const float* __restrict__ sc, const float* __restrict__ v,
    float* __restrict__ out, int T)
{
    int b = blockIdx.x;
    int h = threadIdx.x;
    float m = -INFINITY, den = 0.f, num = 0.f;
    for (int t = 0; t < T; t++) {
        float s  = sc[(size_t)b * T + t];
        float mn = fmaxf(m, s);
        float scale = __expf(m - mn);     // 0 on first step (m = -inf)
        float e     = __expf(s - mn);
        size_t idx = ((size_t)b * T + t) * 256 + h;
        den = den * scale + e;
        num = num * scale + e * v[idx];
        out[idx] = num / den;
        m = mn;
    }
}

// ---------------------------------------------------------------------------
// Host-side launcher
// ---------------------------------------------------------------------------
extern "C" void kernel_launch(void* const* d_in, const int* in_sizes, int n_in,
                              void* d_out, int out_size, void* d_ws, size_t ws_size,
                              hipStream_t stream) {
    const int B = 128, T = 512, D = 128, H = 256;
    const int BT = B * T;                 // 65536
    const int G3 = 3 * H;                 // 768

    const float* x     = (const float*)d_in[0];
    const float* Wp    = (const float*)d_in[1];
    const float* bp    = (const float*)d_in[2];
    const float* Wih_a = (const float*)d_in[3];
    const float* Whh_a = (const float*)d_in[4];
    const float* bih_a = (const float*)d_in[5];
    const float* bhh_a = (const float*)d_in[6];
    const float* Wih_b = (const float*)d_in[7];
    const float* Whh_b = (const float*)d_in[8];
    const float* bih_b = (const float*)d_in[9];
    const float* bhh_b = (const float*)d_in[10];
    const float* Wa    = (const float*)d_in[11];
    const float* ba    = (const float*)d_in[12];
    const float* Wb    = (const float*)d_in[13];
    const float* bb    = (const float*)d_in[14];
    float* out = (float*)d_out;

    // workspace carve-out (256B aligned)
    size_t off = 0;
    auto alloc = [&](size_t bytes) -> char* {
        char* p = (char*)d_ws + off;
        off = (off + bytes + 255) & ~(size_t)255;
        return p;
    };
    bf16*  x_bf    = (bf16*)alloc((size_t)BT * D * sizeof(bf16));
    bf16*  Wp_bf   = (bf16*)alloc((size_t)H * D * sizeof(bf16));
    bf16*  WihA_bf = (bf16*)alloc((size_t)G3 * H * sizeof(bf16));
    bf16*  WhhA_bf = (bf16*)alloc((size_t)G3 * H * sizeof(bf16));
    bf16*  WihB_bf = (bf16*)alloc((size_t)G3 * H * sizeof(bf16));
    bf16*  WhhB_bf = (bf16*)alloc((size_t)G3 * H * sizeof(bf16));
    bf16*  Wb_bf   = (bf16*)alloc((size_t)H * H * sizeof(bf16));
    float* xp_f    = (float*)alloc((size_t)BT * H * sizeof(float));
    bf16*  xp_bf   = (bf16*)alloc((size_t)BT * H * sizeof(bf16));
    bf16*  xWa_bf  = (bf16*)alloc((size_t)BT * G3 * sizeof(bf16));
    bf16*  xWb_bf  = (bf16*)alloc((size_t)BT * G3 * sizeof(bf16));
    bf16*  xa_bf   = (bf16*)alloc((size_t)BT * H * sizeof(bf16));
    bf16*  xb_bf   = (bf16*)alloc((size_t)BT * H * sizeof(bf16));
    float* v_f     = (float*)alloc((size_t)BT * H * sizeof(float));
    float* sc_f    = (float*)alloc((size_t)BT * sizeof(float));

    auto cvt = [&](const float* src, bf16* dst, int n) {
        cvt_bf16_kernel<<<(n + 255) / 256, 256, 0, stream>>>(src, dst, n);
    };
    cvt(x,     x_bf,    BT * D);
    cvt(Wp,    Wp_bf,   H * D);
    cvt(Wih_a, WihA_bf, G3 * H);
    cvt(Whh_a, WhhA_bf, G3 * H);
    cvt(Wih_b, WihB_bf, G3 * H);
    cvt(Whh_b, WhhB_bf, G3 * H);
    cvt(Wb,    Wb_bf,   H * H);

    // xp = x @ Wp^T + bp   (f32 + bf16 out)
    wmma_gemm_kernel<0><<<dim3(BT / 16, H / 64), 32, 0, stream>>>(
        x_bf, Wp_bf, bp, xp_f, xp_bf, nullptr, H, D);

    // xW_{a,b} = xp @ Wih^T + bih  (bf16 out)
    wmma_gemm_kernel<1><<<dim3(BT / 16, G3 / 64), 32, 0, stream>>>(
        xp_bf, WihA_bf, bih_a, nullptr, xWa_bf, nullptr, G3, H);
    wmma_gemm_kernel<1><<<dim3(BT / 16, G3 / 64), 32, 0, stream>>>(
        xp_bf, WihB_bf, bih_b, nullptr, xWb_bf, nullptr, G3, H);

    // sequential GRU scans (independent per 16-row batch slab)
    gru_scan_kernel<<<B / 16, 512, 0, stream>>>(xWa_bf, WhhA_bf, bhh_a, xa_bf, T);
    gru_scan_kernel<<<B / 16, 512, 0, stream>>>(xWb_bf, WhhB_bf, bhh_b, xb_bf, T);

    // v = xp * tanh(xb @ Wb^T + bb)
    wmma_gemm_kernel<2><<<dim3(BT / 16, H / 64), 32, 0, stream>>>(
        xb_bf, Wb_bf, bb, v_f, nullptr, xp_f, H, H);

    // scores + online-softmax causal attention
    scores_kernel<<<BT / 256, 256, 0, stream>>>(xa_bf, Wa, ba, sc_f);
    attn_scan_kernel<<<B, 256, 0, stream>>>(sc_f, v_f, out, T);

    (void)in_sizes; (void)n_in; (void)out_size; (void)ws_size;
}
